// GAT_1168231104585
// MI455X (gfx1250) — matrix operand order repaired and checked by reading, not compiled, not run on hardware
//
#include <hip/hip_runtime.h>
#include <hip/hip_bf16.h>
#include <math.h>

typedef __attribute__((ext_vector_type(16))) _Float16 v16h;
typedef __attribute__((ext_vector_type(8)))  float    v8f;

#define NEG_SLOPE 0.2f

__device__ __forceinline__ void atomicMaxF(float* addr, float v) {
    // ordered-int trick: float order == int order for >=0, reversed-unsigned for <0
    if (v >= 0.0f) atomicMax((int*)addr, __float_as_int(v));
    else           atomicMin((unsigned int*)addr, __float_as_uint(v));
}

// ---------------- W packing into WMMA B-fragment layout (f16) ----------------
// B (32x16 f16) layout: lane n: col = n&15 ; K = (n>>4)*8 + (h&7) + (h>>3)*16  (h = 0..15)
// Stored as [kstep][ntile][lane][16] halves, contiguous.
__global__ void gat_pack_w(const float* __restrict__ W, _Float16* __restrict__ Wp, int K) {
    int idx = blockIdx.x * blockDim.x + threadIdx.x;
    int total = (K >> 5) * 4 * 32 * 16;
    if (idx >= total) return;
    int hh   = idx & 15;
    int lane = (idx >> 4) & 31;
    int nt   = (idx >> 9) & 3;
    int ks   = idx >> 11;
    int col  = nt * 16 + (lane & 15);
    int kk   = ks * 32 + (lane >> 4) * 8 + (hh & 7) + (hh >> 3) * 16;
    Wp[idx] = (_Float16)W[(size_t)kk * 64 + col];
}

// ---------------- GEMM: C[N,64] = A[N,K] * W[K,64], f16 WMMA, f32 accum -----
// block = 128 threads (4 waves); wave w owns col tile w; blockIdx.x = row tile
__global__ __launch_bounds__(128)
void gat_gemm_wmma(const float* __restrict__ A, const _Float16* __restrict__ Wp,
                   float* __restrict__ C, int K, int N) {
    int lane = threadIdx.x & 31;
    int w    = threadIdx.x >> 5;
    int tile = blockIdx.x;
    int row  = tile * 16 + (lane & 15);
    if (row >= N) row = N - 1;          // clamp loads; stores guarded below
    int kb   = (lane >> 4) * 8;         // A frag: halves 0..7 -> K=kb+h ; 8..15 -> K=16+kb+(h-8)
    v8f acc = {};
    int ksteps = K >> 5;
    for (int ks = 0; ks < ksteps; ++ks) {
        const float* ap = A + (size_t)row * K + ks * 32 + kb;
        float4 f0 = *(const float4*)(ap);
        float4 f1 = *(const float4*)(ap + 4);
        float4 f2 = *(const float4*)(ap + 16);
        float4 f3 = *(const float4*)(ap + 20);
        v16h a;
        a[0]=(_Float16)f0.x;  a[1]=(_Float16)f0.y;  a[2]=(_Float16)f0.z;  a[3]=(_Float16)f0.w;
        a[4]=(_Float16)f1.x;  a[5]=(_Float16)f1.y;  a[6]=(_Float16)f1.z;  a[7]=(_Float16)f1.w;
        a[8]=(_Float16)f2.x;  a[9]=(_Float16)f2.y;  a[10]=(_Float16)f2.z; a[11]=(_Float16)f2.w;
        a[12]=(_Float16)f3.x; a[13]=(_Float16)f3.y; a[14]=(_Float16)f3.z; a[15]=(_Float16)f3.w;
        v16h b = *(const v16h*)(Wp + ((size_t)(ks * 4 + w) * 32 + lane) * 16);
        acc = __builtin_amdgcn_wmma_f32_16x16x32_f16(false, a, false, b,
                                                     (short)0, acc, false, false);
    }
    // C/D layout: lanes 0-15 -> M=v, N=lane ; lanes 16-31 -> M=8+v, N=lane-16
    int colbase = w * 16 + (lane & 15);
    int rbase   = tile * 16 + (lane >> 4) * 8;
#pragma unroll
    for (int v = 0; v < 8; ++v) {
        int r = rbase + v;
        if (r < N) C[(size_t)r * 64 + colbase] = acc[v];
    }
}

// ---------------- attention coefficients, layer 1 (8 heads x 8 dims) --------
__global__ void gat_att8(const float* __restrict__ t, const float* __restrict__ as,
                         const float* __restrict__ ad, float* __restrict__ os,
                         float* __restrict__ od, int N) {
    int wid  = (blockIdx.x * blockDim.x + threadIdx.x) >> 5;
    int lane = threadIdx.x & 31;
    if (wid >= N) return;
    float v1 = t[(size_t)wid * 64 + lane];
    float v2 = t[(size_t)wid * 64 + 32 + lane];
    float s1 = v1 * as[lane], s2 = v2 * as[32 + lane];
    float d1 = v1 * ad[lane], d2 = v2 * ad[32 + lane];
#pragma unroll
    for (int m = 1; m <= 4; m <<= 1) {
        s1 += __shfl_xor(s1, m, 32);  s2 += __shfl_xor(s2, m, 32);
        d1 += __shfl_xor(d1, m, 32);  d2 += __shfl_xor(d2, m, 32);
    }
    if ((lane & 7) == 0) {
        int h = lane >> 3;
        os[(size_t)wid * 8 + h]     = s1;
        os[(size_t)wid * 8 + 4 + h] = s2;
        od[(size_t)wid * 8 + h]     = d1;
        od[(size_t)wid * 8 + 4 + h] = d2;
    }
}

// ---------------- attention coefficients, layer 2 (1 head x 64 dims) --------
__global__ void gat_att64(const float* __restrict__ t, const float* __restrict__ as,
                          const float* __restrict__ ad, float* __restrict__ os,
                          float* __restrict__ od, int N) {
    int wid  = (blockIdx.x * blockDim.x + threadIdx.x) >> 5;
    int lane = threadIdx.x & 31;
    if (wid >= N) return;
    float v1 = t[(size_t)wid * 64 + lane];
    float v2 = t[(size_t)wid * 64 + 32 + lane];
    float s = v1 * as[lane] + v2 * as[32 + lane];
    float d = v1 * ad[lane] + v2 * ad[32 + lane];
#pragma unroll
    for (int m = 1; m <= 16; m <<= 1) {
        s += __shfl_xor(s, m, 32);
        d += __shfl_xor(d, m, 32);
    }
    if (lane == 0) { os[wid] = s; od[wid] = d; }
}

// ---------------- init kernels ----------------------------------------------
__global__ void gat_init_l1(float* __restrict__ agg, const float* __restrict__ b,
                            float* __restrict__ m, float* __restrict__ den, int N) {
    int idx = blockIdx.x * blockDim.x + threadIdx.x;
    if (idx >= N * 64) return;
    agg[idx] = b[idx & 63];
    int c = idx & 63;
    if (c < 8) {
        size_t p = (size_t)(idx >> 6) * 8 + c;
        m[p]   = __int_as_float(0xff800000u); // -inf
        den[p] = 0.0f;
    }
}

__global__ void gat_init_l2(float* __restrict__ out, const float* __restrict__ b,
                            float* __restrict__ m, float* __restrict__ den, int N) {
    int idx = blockIdx.x * blockDim.x + threadIdx.x;
    if (idx >= N * 64) return;
    out[idx] = b[idx & 63];
    if ((idx & 63) == 0) {
        m[idx >> 6]   = __int_as_float(0xff800000u);
        den[idx >> 6] = 0.0f;
    }
}

// ---------------- edge passes, layer 1 (per (edge, head)) -------------------
__device__ __forceinline__ void edge_sd(const long long* ei, long long e,
                                        long long E, int& s, int& d) {
    if (e < E) { s = (int)ei[e]; d = (int)ei[E + e]; }
    else       { s = d = (int)(e - E); }
}

__global__ void gat_edge_max8(const long long* __restrict__ ei, const float* __restrict__ as,
                              const float* __restrict__ ad, float* __restrict__ m,
                              long long E, long long Et) {
    long long idx = (long long)blockIdx.x * blockDim.x + threadIdx.x;
    if (idx >= Et * 8) return;
    long long e = idx >> 3; int hd = (int)(idx & 7);
    int s, d; edge_sd(ei, e, E, s, d);
    float z = as[(size_t)s * 8 + hd] + ad[(size_t)d * 8 + hd];
    float l = z > 0.0f ? z : NEG_SLOPE * z;
    atomicMaxF(&m[(size_t)d * 8 + hd], l);
}

__global__ void gat_edge_sum8(const long long* __restrict__ ei, const float* __restrict__ as,
                              const float* __restrict__ ad, const float* __restrict__ m,
                              float* __restrict__ den, long long E, long long Et) {
    long long idx = (long long)blockIdx.x * blockDim.x + threadIdx.x;
    if (idx >= Et * 8) return;
    long long e = idx >> 3; int hd = (int)(idx & 7);
    int s, d; edge_sd(ei, e, E, s, d);
    float z = as[(size_t)s * 8 + hd] + ad[(size_t)d * 8 + hd];
    float l = z > 0.0f ? z : NEG_SLOPE * z;
    atomicAdd(&den[(size_t)d * 8 + hd], expf(l - m[(size_t)d * 8 + hd]));
}

// wave per edge: lanes cover 64 channels (2 each)
__global__ void gat_edge_agg8(const long long* __restrict__ ei, const float* __restrict__ as,
                              const float* __restrict__ ad, const float* __restrict__ m,
                              const float* __restrict__ den, const float* __restrict__ feat,
                              float* __restrict__ agg, long long E, long long Et) {
    long long wid = ((long long)blockIdx.x * blockDim.x + threadIdx.x) >> 5;
    int lane = threadIdx.x & 31;
    if (wid >= Et) return;
    int s, d; edge_sd(ei, wid, E, s, d);
#pragma unroll
    for (int j = 0; j < 2; ++j) {
        int c = lane + 32 * j;
        int hd = c >> 3;
        float z = as[(size_t)s * 8 + hd] + ad[(size_t)d * 8 + hd];
        float l = z > 0.0f ? z : NEG_SLOPE * z;
        float alpha = expf(l - m[(size_t)d * 8 + hd]) / den[(size_t)d * 8 + hd];
        atomicAdd(&agg[(size_t)d * 64 + c], feat[(size_t)s * 64 + c] * alpha);
    }
}

// ---------------- ELU --------------------------------------------------------
__global__ void gat_elu(float* __restrict__ a, int n) {
    int idx = blockIdx.x * blockDim.x + threadIdx.x;
    if (idx >= n) return;
    float v = a[idx];
    a[idx] = v > 0.0f ? v : expm1f(v);
}

// ---------------- edge passes, layer 2 (scalar per edge) --------------------
__global__ void gat_edge_max1(const long long* __restrict__ ei, const float* __restrict__ as,
                              const float* __restrict__ ad, float* __restrict__ m,
                              long long E, long long Et) {
    long long e = (long long)blockIdx.x * blockDim.x + threadIdx.x;
    if (e >= Et) return;
    int s, d; edge_sd(ei, e, E, s, d);
    float z = as[s] + ad[d];
    float l = z > 0.0f ? z : NEG_SLOPE * z;
    atomicMaxF(&m[d], l);
}

__global__ void gat_edge_sum1(const long long* __restrict__ ei, const float* __restrict__ as,
                              const float* __restrict__ ad, const float* __restrict__ m,
                              float* __restrict__ den, long long E, long long Et) {
    long long e = (long long)blockIdx.x * blockDim.x + threadIdx.x;
    if (e >= Et) return;
    int s, d; edge_sd(ei, e, E, s, d);
    float z = as[s] + ad[d];
    float l = z > 0.0f ? z : NEG_SLOPE * z;
    atomicAdd(&den[d], expf(l - m[d]));
}

__global__ void gat_edge_agg1(const long long* __restrict__ ei, const float* __restrict__ as,
                              const float* __restrict__ ad, const float* __restrict__ m,
                              const float* __restrict__ den, const float* __restrict__ feat,
                              float* __restrict__ out, long long E, long long Et) {
    long long wid = ((long long)blockIdx.x * blockDim.x + threadIdx.x) >> 5;
    int lane = threadIdx.x & 31;
    if (wid >= Et) return;
    int s, d; edge_sd(ei, wid, E, s, d);
    float z = as[s] + ad[d];
    float l = z > 0.0f ? z : NEG_SLOPE * z;
    float alpha = expf(l - m[d]) / den[d];
#pragma unroll
    for (int j = 0; j < 2; ++j) {
        int c = lane + 32 * j;
        atomicAdd(&out[(size_t)d * 64 + c], feat[(size_t)s * 64 + c] * alpha);
    }
}

// ============================================================================
extern "C" void kernel_launch(void* const* d_in, const int* in_sizes, int n_in,
                              void* d_out, int out_size, void* d_ws, size_t ws_size,
                              hipStream_t stream) {
    const float*     x   = (const float*)d_in[0];
    const long long* ei  = (const long long*)d_in[1];
    const float*     W1  = (const float*)d_in[2];
    const float*     as1 = (const float*)d_in[3];   // [8,8] flat 64
    const float*     ad1 = (const float*)d_in[4];
    const float*     b1  = (const float*)d_in[5];
    const float*     W2  = (const float*)d_in[6];
    const float*     as2 = (const float*)d_in[7];   // [1,64]
    const float*     ad2 = (const float*)d_in[8];
    const float*     b2  = (const float*)d_in[9];
    float*           out = (float*)d_out;

    const int       N  = in_sizes[0] / 512;
    const long long E  = (long long)in_sizes[1] / 2;
    const long long Et = E + N;

    // workspace layout (all region sizes multiple of 32B)
    char* w = (char*)d_ws;
    float* h1   = (float*)w; w += (size_t)N * 64 * 4;   // layer1 features, reused as h2
    float* agg1 = (float*)w; w += (size_t)N * 64 * 4;   // layer1 aggregate -> ELU -> layer2 input
    float* aS   = (float*)w; w += (size_t)N * 8 * 4;
    float* aD   = (float*)w; w += (size_t)N * 8 * 4;
    float* mx   = (float*)w; w += (size_t)N * 8 * 4;
    float* den  = (float*)w; w += (size_t)N * 8 * 4;
    _Float16* wp1 = (_Float16*)w; w += 512 * 64 * 2;
    _Float16* wp2 = (_Float16*)w; w += 64 * 64 * 2;

    const int B = 256;
    int rowTiles = (N + 15) / 16;

    // ---- weight packing ----
    {
        int t1 = (512 / 32) * 4 * 32 * 16;
        gat_pack_w<<<(t1 + B - 1) / B, B, 0, stream>>>(W1, wp1, 512);
        int t2 = (64 / 32) * 4 * 32 * 16;
        gat_pack_w<<<(t2 + B - 1) / B, B, 0, stream>>>(W2, wp2, 64);
    }

    // ================= layer 1 =================
    gat_gemm_wmma<<<rowTiles, 128, 0, stream>>>(x, wp1, h1, 512, N);

    {
        long long t = (long long)N * 32;
        gat_att8<<<(int)((t + B - 1) / B), B, 0, stream>>>(h1, as1, ad1, aS, aD, N);
    }
    {
        long long t = (long long)N * 64;
        gat_init_l1<<<(int)((t + B - 1) / B), B, 0, stream>>>(agg1, b1, mx, den, N);
    }
    {
        long long t = Et * 8;
        gat_edge_max8<<<(int)((t + B - 1) / B), B, 0, stream>>>(ei, aS, aD, mx, E, Et);
        gat_edge_sum8<<<(int)((t + B - 1) / B), B, 0, stream>>>(ei, aS, aD, mx, den, E, Et);
    }
    {
        long long t = Et * 32;
        gat_edge_agg8<<<(int)((t + B - 1) / B), B, 0, stream>>>(ei, aS, aD, mx, den, h1, agg1, E, Et);
    }
    {
        long long t = (long long)N * 64;
        gat_elu<<<(int)((t + B - 1) / B), B, 0, stream>>>(agg1, N * 64);
    }

    // ================= layer 2 =================
    float* h2 = h1;                 // reuse
    float* aS2 = aS, *aD2 = aD;     // reuse (only N entries used)
    float* mx2 = mx, *den2 = den;   // reuse

    gat_gemm_wmma<<<rowTiles, 128, 0, stream>>>(agg1, wp2, h2, 64, N);

    {
        long long t = (long long)N * 32;
        gat_att64<<<(int)((t + B - 1) / B), B, 0, stream>>>(h2, as2, ad2, aS2, aD2, N);
    }
    {
        long long t = (long long)N * 64;
        gat_init_l2<<<(int)((t + B - 1) / B), B, 0, stream>>>(out, b2, mx2, den2, N);
    }
    gat_edge_max1<<<(int)((Et + B - 1) / B), B, 0, stream>>>(ei, aS2, aD2, mx2, E, Et);
    gat_edge_sum1<<<(int)((Et + B - 1) / B), B, 0, stream>>>(ei, aS2, aD2, mx2, den2, E, Et);
    {
        long long t = Et * 32;
        gat_edge_agg1<<<(int)((t + B - 1) / B), B, 0, stream>>>(ei, aS2, aD2, mx2, den2, h2, out, E, Et);
    }
}
